// L2XModel_79577154060806
// MI455X (gfx1250) — matrix operand order repaired and checked by reading, not compiled, not run on hardware
//
#include <hip/hip_runtime.h>
#include <hip/hip_bf16.h>

typedef __attribute__((ext_vector_type(16))) _Float16 v16h;
typedef __attribute__((ext_vector_type(8)))  float    v8f;

#define EPS_F32 1.1920928955078125e-07f

// ---------------------------------------------------------------------------
// Prep: convert f32 weights into padded f16, pre-swizzled into WMMA B-fragment
// layout: frag[(kc*NT + nt)*32 + lane][16 halves], where for lane L:
//   n = nt*16 + (L&15),  K = kc*32 + ((L>=16)?8:0) + j + ((j>=8)?8:0)   (j = 0..15)
// One thread per (tile, lane): writes 16 contiguous halves (32 B).
//   w1: 200x100 -> KT=7 (K=224), NT=7  (N=112)  : 49 tiles
//   w2: 100x100 -> KT=4 (K=128), NT=7  (N=112)  : 28 tiles
//   wl: 100x200 -> KT=4 (K=128), NT=13 (N=208)  : 52 tiles
// ---------------------------------------------------------------------------
__global__ __launch_bounds__(256) void prep_weights(
    const float* __restrict__ w1, const float* __restrict__ w2,
    const float* __restrict__ wl,
    _Float16* __restrict__ w1f, _Float16* __restrict__ w2f,
    _Float16* __restrict__ wlf)
{
  int t = blockIdx.x * blockDim.x + threadIdx.x;
  const float* src; _Float16* dst; int NT, fanin, fanout;
  if (t < 49 * 32)        { src = w1; dst = w1f; NT = 7;  fanin = 200; fanout = 100; }
  else if (t < 77 * 32)   { t -= 49 * 32; src = w2; dst = w2f; NT = 7;  fanin = 100; fanout = 100; }
  else if (t < 129 * 32)  { t -= 77 * 32; src = wl; dst = wlf; NT = 13; fanin = 100; fanout = 200; }
  else return;

  int lane = t & 31;
  int tile = t >> 5;
  int kc   = tile / NT;
  int nt   = tile % NT;
  int n    = nt * 16 + (lane & 15);
  int kh   = (lane >= 16) ? 8 : 0;
  _Float16* out = dst + (size_t)(tile * 32 + lane) * 16;
#pragma unroll
  for (int j = 0; j < 16; ++j) {
    int k = kc * 32 + kh + j + ((j >= 8) ? 8 : 0);
    float v = (k < fanin && n < fanout) ? src[k * fanout + n] : 0.0f;
    out[j] = (_Float16)v;
  }
}

// ---------------------------------------------------------------------------
// Fused L2X forward: per block = 16 rows.
//   Stage A (WMMA): h1 = relu(x@w1+b1); h2 = relu(h1@w2+b2); logits = h2@wl+bl
//   Stage B (wave/row): samples = max_k softmax((gumbel + logits)/tau)
//                       preds   = softmax((x*samples)@wo + bo)
// ---------------------------------------------------------------------------
__global__ __launch_bounds__(256) void l2x_fused(
    const float* __restrict__ x, const float* __restrict__ uniform,
    const float* __restrict__ b1, const float* __restrict__ b2,
    const float* __restrict__ bl, const float* __restrict__ wo,
    const float* __restrict__ bo,
    const _Float16* __restrict__ w1f, const _Float16* __restrict__ w2f,
    const _Float16* __restrict__ wlf,
    float* __restrict__ preds_out, float* __restrict__ samples_out)
{
  __shared__ __align__(16) float    xs_tile[16 * 200];   // x tile f32 (for xs later)
  __shared__ __align__(16) _Float16 xh_tile[16 * 224];   // x tile f16, K-padded (GEMM1 A)
  __shared__ __align__(16) _Float16 h1s[16 * 128];       // hidden 1 (f16, K-padded)
  __shared__ __align__(16) _Float16 h2s[16 * 128];       // hidden 2 (f16, K-padded)
  __shared__ __align__(16) float    logit_s[16 * 200];   // logits (f32)

  const int tid    = threadIdx.x;
  const int lane   = tid & 31;
  const int wave   = tid >> 5;
  const int row0   = blockIdx.x * 16;
  const int half16 = (lane >= 16) ? 1 : 0;   // which half-wave
  const int m      = lane & 15;              // A-fragment row / C-fragment col

  // Zero K-padding of hidden activations (cols 112..127 never written otherwise)
  for (int i = tid; i < 16 * 128; i += 256) { h1s[i] = (_Float16)0.0f; h2s[i] = (_Float16)0.0f; }

  // Cooperative x-tile load: 16 rows x 200 f32 = one contiguous 3200-float span
  {
    const float4* xg = (const float4*)(x + (size_t)row0 * 200);
    float4* xl = (float4*)xs_tile;
    for (int i = tid; i < 800; i += 256) xl[i] = xg[i];
  }
  __syncthreads();

  // f32 -> f16 x-tile with zero K-padding (one pass; A-frags become pure b128 loads)
  for (int i = tid; i < 16 * 224; i += 256) {
    int row = i / 224;
    int k   = i - row * 224;
    float v = (k < 200) ? xs_tile[row * 200 + k] : 0.0f;
    xh_tile[i] = (_Float16)v;
  }
  __syncthreads();

  // ---- GEMM1: h1[16x112] = relu(x[16x224] @ w1f[224x112] + b1), waves 0..6 ----
  if (wave < 7) {
    v8f acc = {};
    for (int kc = 0; kc < 7; ++kc) {
      const int kb = kc * 32 + half16 * 8;   // two contiguous 8-half runs: kb, kb+16
      v16h a;
#pragma unroll
      for (int j = 0; j < 8; ++j)  a[j] = xh_tile[m * 224 + kb + j];
#pragma unroll
      for (int j = 8; j < 16; ++j) a[j] = xh_tile[m * 224 + kb + 8 + j];
      v16h b = *(const v16h*)(w1f + (size_t)((kc * 7 + wave) * 32 + lane) * 16);
      acc = __builtin_amdgcn_wmma_f32_16x16x32_f16(false, a, false, b, (short)0, acc, false, false);
    }
    int nb = wave * 16 + (lane & 15);
    int mr = half16 * 8;
    float bias = (nb < 100) ? b1[nb] : 0.0f;
#pragma unroll
    for (int r = 0; r < 8; ++r) {
      float v = (nb < 100) ? fmaxf(acc[r] + bias, 0.0f) : 0.0f;
      h1s[(mr + r) * 128 + nb] = (_Float16)v;
    }
  }
  __syncthreads();

  // ---- GEMM2: h2[16x112] = relu(h1[16x128] @ w2f[128x112] + b2), waves 0..6 ----
  if (wave < 7) {
    v8f acc = {};
    for (int kc = 0; kc < 4; ++kc) {
      const int kb = kc * 32 + half16 * 8;
      v16h a;
#pragma unroll
      for (int j = 0; j < 8; ++j)  a[j] = h1s[m * 128 + kb + j];
#pragma unroll
      for (int j = 8; j < 16; ++j) a[j] = h1s[m * 128 + kb + 8 + j];
      v16h b = *(const v16h*)(w2f + (size_t)((kc * 7 + wave) * 32 + lane) * 16);
      acc = __builtin_amdgcn_wmma_f32_16x16x32_f16(false, a, false, b, (short)0, acc, false, false);
    }
    int nb = wave * 16 + (lane & 15);
    int mr = half16 * 8;
    float bias = (nb < 100) ? b2[nb] : 0.0f;
#pragma unroll
    for (int r = 0; r < 8; ++r) {
      float v = (nb < 100) ? fmaxf(acc[r] + bias, 0.0f) : 0.0f;
      h2s[(mr + r) * 128 + nb] = (_Float16)v;
    }
  }
  __syncthreads();

  // ---- GEMM3: logits[16x208] = h2[16x128] @ wlf[128x208] + bl, 13 N-tiles / 8 waves ----
  for (int nt = wave; nt < 13; nt += 8) {
    v8f acc = {};
    for (int kc = 0; kc < 4; ++kc) {
      const int kb = kc * 32 + half16 * 8;
      v16h a;
#pragma unroll
      for (int j = 0; j < 8; ++j)  a[j] = h2s[m * 128 + kb + j];
#pragma unroll
      for (int j = 8; j < 16; ++j) a[j] = h2s[m * 128 + kb + 8 + j];
      v16h b = *(const v16h*)(wlf + (size_t)((kc * 13 + nt) * 32 + lane) * 16);
      acc = __builtin_amdgcn_wmma_f32_16x16x32_f16(false, a, false, b, (short)0, acc, false, false);
    }
    int nb = nt * 16 + (lane & 15);
    if (nb < 200) {
      float bias = bl[nb];
      int mr = half16 * 8;
#pragma unroll
      for (int r = 0; r < 8; ++r) logit_s[(mr + r) * 200 + nb] = acc[r] + bias;
    }
  }
  __syncthreads();

  // ---- Sampling: one wave per row, rows {2*wave, 2*wave+1} ----
  for (int rr = 0; rr < 2; ++rr) {
    const int r = wave * 2 + rr;
    const size_t row = (size_t)(row0 + r);

    float lg[7], smax[7];
#pragma unroll
    for (int i = 0; i < 7; ++i) {
      int n = lane + 32 * i;
      lg[i] = (n < 200) ? logit_s[r * 200 + n] * 2.0f : -__builtin_inff();  // 1/tau = 2
      smax[i] = 0.0f;
    }

    for (int k = 0; k < 10; ++k) {
      const float* up = uniform + (row * 10 + (size_t)k) * 200;
      if (k < 9) __builtin_prefetch(up + 200 + lane * 32, 0, 0);  // stream next sample row

      float z[7];
      float zmax = -__builtin_inff();
#pragma unroll
      for (int i = 0; i < 7; ++i) {
        int n = lane + 32 * i;
        if (n < 200) {
          float u = fminf(fmaxf(up[n], EPS_F32), 1.0f);
          float g = -__logf(-__logf(u));          // gumbel
          z[i] = g * 2.0f + lg[i];                // (g + logits)/tau
          zmax = fmaxf(zmax, z[i]);
        } else {
          z[i] = -__builtin_inff();
        }
      }
      for (int off = 16; off >= 1; off >>= 1) zmax = fmaxf(zmax, __shfl_xor(zmax, off, 32));

      float e[7], sum = 0.0f;
#pragma unroll
      for (int i = 0; i < 7; ++i) { e[i] = __expf(z[i] - zmax); sum += e[i]; }
      for (int off = 16; off >= 1; off >>= 1) sum += __shfl_xor(sum, off, 32);

      float inv = 1.0f / sum;
#pragma unroll
      for (int i = 0; i < 7; ++i) smax[i] = fmaxf(smax[i], e[i] * inv);
    }

    // samples out, xs = x*samples, preds = softmax(xs@wo + bo)
    float s0 = 0.0f, s1 = 0.0f;
#pragma unroll
    for (int i = 0; i < 7; ++i) {
      int n = lane + 32 * i;
      if (n < 200) {
        float s = smax[i];
        samples_out[row * 200 + n] = s;
        float xsv = xs_tile[r * 200 + n] * s;
        s0 += xsv * wo[2 * n];
        s1 += xsv * wo[2 * n + 1];
      }
    }
    for (int off = 16; off >= 1; off >>= 1) {
      s0 += __shfl_xor(s0, off, 32);
      s1 += __shfl_xor(s1, off, 32);
    }
    if (lane == 0) {
      float l0 = s0 + bo[0], l1 = s1 + bo[1];
      float mx = fmaxf(l0, l1);
      float e0 = __expf(l0 - mx), e1 = __expf(l1 - mx);
      float inv = 1.0f / (e0 + e1);
      preds_out[row * 2 + 0] = e0 * inv;
      preds_out[row * 2 + 1] = e1 * inv;
    }
  }
}

// ---------------------------------------------------------------------------
extern "C" void kernel_launch(void* const* d_in, const int* in_sizes, int n_in,
                              void* d_out, int out_size, void* d_ws, size_t ws_size,
                              hipStream_t stream) {
  const float* x       = (const float*)d_in[0];  // (32768, 200)
  const float* uniform = (const float*)d_in[1];  // (32768, 10, 200)
  const float* w1      = (const float*)d_in[2];  // (200, 100)
  const float* b1      = (const float*)d_in[3];  // (100,)
  const float* w2      = (const float*)d_in[4];  // (100, 100)
  const float* b2      = (const float*)d_in[5];  // (100,)
  const float* wl      = (const float*)d_in[6];  // (100, 200)
  const float* bl      = (const float*)d_in[7];  // (200,)
  const float* wo      = (const float*)d_in[8];  // (200, 2)
  const float* bo      = (const float*)d_in[9];  // (2,)

  float* preds   = (float*)d_out;                // (32768, 2)
  float* samples = (float*)d_out + 32768 * 2;    // (32768, 200)

  // Workspace: pre-swizzled f16 B-fragments (49+28+52 tiles * 512 halves)
  _Float16* w1f = (_Float16*)d_ws;               // 49*512 = 25088 halves
  _Float16* w2f = w1f + 49 * 512;                // 28*512 = 14336 halves
  _Float16* wlf = w2f + 28 * 512;                // 52*512 = 26624 halves

  prep_weights<<<(129 * 32 + 255) / 256, 256, 0, stream>>>(w1, w2, wl, w1f, w2f, wlf);
  l2x_fused<<<32768 / 16, 256, 0, stream>>>(x, uniform, b1, b2, bl, wo, bo,
                                            w1f, w2f, wlf, preds, samples);
}
